// TemporalAttention_80968723464418
// MI455X (gfx1250) — matrix operand order repaired and checked by reading, not compiled, not run on hardware
//
#include <hip/hip_runtime.h>

typedef __bf16 bf16_t;
typedef __attribute__((ext_vector_type(16))) __bf16 v16bf;
typedef __attribute__((ext_vector_type(8)))  float  v8f;

#define C_IN   512
#define C_QKV  1536
#define S_TOK  16384   // t*h*w per batch
#define N_TOK  32768

static __device__ __forceinline__ bf16_t to_bf16(float f) { return (bf16_t)f; }

static __device__ __forceinline__ v8f zero8() {
  v8f z;
#pragma unroll
  for (int i = 0; i < 8; ++i) z[i] = 0.0f;
  return z;
}

// 32-bit LDS byte offset (what GLOBAL_LOAD_ASYNC_TO_LDS wants in its VDST VGPR)
static __device__ __forceinline__ unsigned lds_addr_of(const void* p) {
  return (unsigned)(unsigned long long)(__attribute__((address_space(3))) const char*)p;
}

// async copy 16B global -> LDS (ASYNCcnt-tracked, no VGPR round trip)
static __device__ __forceinline__ void async_ld_b128(unsigned lds_addr, const void* gptr) {
  asm volatile("global_load_async_to_lds_b128 %0, %1, off"
               :: "v"(lds_addr), "v"((unsigned long long)gptr)
               : "memory");
}

#define WAIT_ASYNC(n) asm volatile("s_wait_asynccnt %0" :: "i"(n) : "memory")

// A fragment (16x32 bf16). Lanes 0-15: row M=lane, K=0..7 (v0-3) and K=16..23 (v4-7);
// lanes 16-31: K=8..15 and 24..31.
static __device__ __forceinline__ v16bf load_a_frag(const bf16_t* __restrict__ base, int ld, int lane) {
  const int row = lane & 15;
  const int kb  = (lane >> 4) << 3;          // 0 or 8
  const bf16_t* p = base + row * ld + kb;
  v16bf a;
#pragma unroll
  for (int i = 0; i < 8; ++i) { a[i] = p[i]; a[i + 8] = p[16 + i]; }
  return a;
}

// B fragment (32x16 bf16). Column N=lane&15; lanes 0-15 hold K=0..7,16..23, lanes 16-31 K=8..15,24..31.
// `base` row n of the array = column n of B, contiguous in K.
static __device__ __forceinline__ v16bf load_b_frag(const bf16_t* __restrict__ base, int ld, int lane) {
  const int col = lane & 15;
  const int ks  = (lane >> 4) << 3;          // 0 or 8
  const bf16_t* p = base + col * ld + ks;
  v16bf b;
#pragma unroll
  for (int i = 0; i < 8; ++i) { b[i] = p[i]; b[i + 8] = p[16 + i]; }
  return b;
}

// ---------------- x (b,c,s) f32  ->  xt (b*s, c) bf16, LDS tile transpose ----------------
__global__ __launch_bounds__(256) void xpose_kernel(const float* __restrict__ x,
                                                    bf16_t* __restrict__ xt) {
  __shared__ float tile[32][33];
  const int s0 = blockIdx.x * 32;
  const int c0 = blockIdx.y * 32;
  const int b  = blockIdx.z;
  const int tx = threadIdx.x, ty = threadIdx.y;
#pragma unroll
  for (int j = 0; j < 32; j += 8)
    tile[ty + j][tx] = x[((size_t)(b * C_IN + c0 + ty + j)) * S_TOK + s0 + tx];
  __syncthreads();
#pragma unroll
  for (int j = 0; j < 32; j += 8)
    xt[((size_t)(b * S_TOK + s0 + ty + j)) * C_IN + c0 + tx] = to_bf16(tile[tx][ty + j]);
}

__global__ void cvt_kernel(const float* __restrict__ src, bf16_t* __restrict__ dst, int n) {
  int i = blockIdx.x * blockDim.x + threadIdx.x;
  if (i < n) dst[i] = to_bf16(src[i]);
}

// =====================  double-buffered async WMMA GEMM core  =====================
// BM=128, BN=128, BK=32; 8 waves in 4x2 grid, each computing 32x64 (2x4 frags).
// Next K-tile streamed into the alternate LDS buffer with GLOBAL_LOAD_ASYNC_TO_LDS_B128
// while WMMAs consume the current one; sync with s_wait_asynccnt + barrier.

// ---------------- QKV GEMM: [N,512] x [1536,512]^T + bias -> bf16 [N,1536] ----------------
__global__ __launch_bounds__(256) void qkv_gemm_kernel(const bf16_t* __restrict__ A,
                                                       const bf16_t* __restrict__ W,
                                                       const float*  __restrict__ bias,
                                                       bf16_t* __restrict__ out) {
  __shared__ bf16_t As[2][128 * 32];
  __shared__ bf16_t Bs[2][128 * 32];
  const int tid = threadIdx.x, lane = tid & 31, wv = tid >> 5;
  const int wm = wv >> 1, wn = wv & 1;              // 4x2 wave grid, 32x64 per wave
  const int rowBase = blockIdx.y * 128;
  const int colBase = blockIdx.x * 128;

  // per-thread copy slice: 16 bf16 of one row per operand (two b128 asyncs each)
  const int crow = tid >> 1;                 // 0..127
  const int ccol = (tid & 1) << 4;           // 0 or 16
  const bf16_t* gA = A + (size_t)(rowBase + crow) * C_IN + ccol;
  const bf16_t* gB = W + (size_t)(colBase + crow) * C_IN + ccol;
  const unsigned lA = lds_addr_of(&As[0][crow * 32 + ccol]);
  const unsigned lB = lds_addr_of(&Bs[0][crow * 32 + ccol]);

  // prologue: stream K-tile 0 into buffer 0
  async_ld_b128(lA,      gA);
  async_ld_b128(lA + 16, gA + 8);
  async_ld_b128(lB,      gB);
  async_ld_b128(lB + 16, gB + 8);

  v8f acc[2][4];
#pragma unroll
  for (int i = 0; i < 2; ++i)
#pragma unroll
    for (int j = 0; j < 4; ++j) acc[i][j] = zero8();

  for (int k = 0; k < C_IN / 32; ++k) {
    const int p = k & 1;
    if (k + 1 < C_IN / 32) {                 // stream next tile into other buffer
      const unsigned boff = (unsigned)((k + 1) & 1) * (128 * 32 * 2);
      const bf16_t* nA = gA + (k + 1) * 32;
      const bf16_t* nB = gB + (k + 1) * 32;
      async_ld_b128(lA + boff,      nA);
      async_ld_b128(lA + boff + 16, nA + 8);
      async_ld_b128(lB + boff,      nB);
      async_ld_b128(lB + boff + 16, nB + 8);
      WAIT_ASYNC(4);                         // current tile (older 4) complete
    } else {
      WAIT_ASYNC(0);
    }
    __syncthreads();
#pragma unroll
    for (int mf = 0; mf < 2; ++mf) {
      v16bf a = load_a_frag(&As[p][(wm * 32 + mf * 16) * 32], 32, lane);
#pragma unroll
      for (int nf = 0; nf < 4; ++nf) {
        v16bf b = load_b_frag(&Bs[p][(wn * 64 + nf * 16) * 32], 32, lane);
        acc[mf][nf] = __builtin_amdgcn_wmma_f32_16x16x32_bf16(false, a, false, b,
                                                              (short)0, acc[mf][nf], false, false);
      }
    }
    __syncthreads();                         // all waves done with buffer p
  }

  const int hi = lane >> 4, cl = lane & 15;
#pragma unroll
  for (int mf = 0; mf < 2; ++mf)
#pragma unroll
    for (int nf = 0; nf < 4; ++nf) {
      const int col = colBase + wn * 64 + nf * 16 + cl;
      const float bv = bias[col];
      const int row0 = rowBase + wm * 32 + mf * 16 + hi * 8;
#pragma unroll
      for (int i = 0; i < 8; ++i)
        out[(size_t)(row0 + i) * C_QKV + col] = to_bf16(acc[mf][nf][i] + bv);
    }
}

// ------------- per-token attention: attn = softmax(q^T k / 8); O = attn @ v^T -------------
__global__ __launch_bounds__(64) void attn_kernel(const bf16_t* __restrict__ qkv,
                                                  bf16_t* __restrict__ aout) {
  __shared__ bf16_t qt[2][64 * 32];   // q^T [d][hk], K padded 8->32 with zeros
  __shared__ bf16_t kt[2][64 * 32];   // k^T [e][hk]
  __shared__ bf16_t vl[2][16 * 64];   // v   [h][e], h padded 8->16 with zeros
  __shared__ bf16_t at[2][64 * 64];   // softmaxed attn, bf16

  const int lane = threadIdx.x & 31;
  const int wv   = threadIdx.x >> 5;
  const int n    = blockIdx.x * 2 + wv;
  const bf16_t* base = qkv + (size_t)n * C_QKV;
  bf16_t* QT = qt[wv]; bf16_t* KT = kt[wv]; bf16_t* VL = vl[wv]; bf16_t* AT = at[wv];

  {
    const int hk = lane;                       // one K column per lane
    const bool valid = hk < 8;
#pragma unroll 4
    for (int d = 0; d < 64; ++d) {
      QT[d * 32 + hk] = valid ? base[hk * 64 + d]       : (bf16_t)0.0f;
      KT[d * 32 + hk] = valid ? base[512 + hk * 64 + d] : (bf16_t)0.0f;
    }
  }
  for (int idx = lane; idx < 16 * 64; idx += 32)
    VL[idx] = ((idx >> 6) < 8) ? base[1024 + idx] : (bf16_t)0.0f;
  __syncthreads();

  // stage 1: attn[d][e] = sum_h q[h][d] * k[h][e]   (16 WMMAs, 4x4 frags)
  v8f att[4][4];
#pragma unroll
  for (int fm = 0; fm < 4; ++fm) {
    v16bf a = load_a_frag(&QT[fm * 16 * 32], 32, lane);
#pragma unroll
    for (int fn = 0; fn < 4; ++fn) {
      v16bf b = load_b_frag(&KT[fn * 16 * 32], 32, lane);
      att[fm][fn] = __builtin_amdgcn_wmma_f32_16x16x32_bf16(false, a, false, b,
                                                            (short)0, zero8(), false, false);
    }
  }

  // softmax over e: each row of a 16x16 C frag lives in one 16-lane half -> xor-shuffle reduce
  const float scale = 0.125f;   // D^-0.5 = 1/8
  const int hi = lane >> 4, cl = lane & 15;
#pragma unroll
  for (int fm = 0; fm < 4; ++fm) {
#pragma unroll
    for (int i = 0; i < 8; ++i) {
      float v[4]; float m = -3.0e38f;
#pragma unroll
      for (int fn = 0; fn < 4; ++fn) { v[fn] = att[fm][fn][i] * scale; m = fmaxf(m, v[fn]); }
#pragma unroll
      for (int off = 1; off < 16; off <<= 1) m = fmaxf(m, __shfl_xor(m, off, 32));
      float s = 0.0f;
#pragma unroll
      for (int fn = 0; fn < 4; ++fn) { v[fn] = __expf(v[fn] - m); s += v[fn]; }
#pragma unroll
      for (int off = 1; off < 16; off <<= 1) s += __shfl_xor(s, off, 32);
      const float inv = 1.0f / s;
      const int row = fm * 16 + i + hi * 8;
#pragma unroll
      for (int fn = 0; fn < 4; ++fn)
        AT[row * 64 + fn * 16 + cl] = to_bf16(v[fn] * inv);
    }
  }
  __syncthreads();

  // stage 2: O[d][h] = sum_e attn[d][e] * v[h][e]   (8 WMMAs)
#pragma unroll
  for (int fm = 0; fm < 4; ++fm) {
    v8f o = zero8();
#pragma unroll
    for (int kk = 0; kk < 2; ++kk) {
      v16bf a = load_a_frag(&AT[fm * 16 * 64 + kk * 32], 64, lane);
      v16bf b = load_b_frag(&VL[kk * 32], 64, lane);
      o = __builtin_amdgcn_wmma_f32_16x16x32_bf16(false, a, false, b, (short)0, o, false, false);
    }
    if (cl < 8) {
#pragma unroll
      for (int i = 0; i < 8; ++i) {
        const int d = fm * 16 + i + hi * 8;
        aout[(size_t)n * C_IN + cl * 64 + d] = to_bf16(o[i]);   // out[n][h*64+d]
      }
    }
  }
}

// ------- proj GEMM: [N,512] x [512,512]^T + bias -> f32, stored transposed (b,c,s) -------
__global__ __launch_bounds__(256) void proj_gemm_kernel(const bf16_t* __restrict__ A,
                                                        const bf16_t* __restrict__ W,
                                                        const float*  __restrict__ bias,
                                                        float* __restrict__ out) {
  __shared__ bf16_t As[2][128 * 32];
  __shared__ bf16_t Bs[2][128 * 32];
  const int tid = threadIdx.x, lane = tid & 31, wv = tid >> 5;
  const int wm = wv >> 1, wn = wv & 1;
  const int rowBase = blockIdx.y * 128;
  const int colBase = blockIdx.x * 128;

  const int crow = tid >> 1;
  const int ccol = (tid & 1) << 4;
  const bf16_t* gA = A + (size_t)(rowBase + crow) * C_IN + ccol;
  const bf16_t* gB = W + (size_t)(colBase + crow) * C_IN + ccol;
  const unsigned lA = lds_addr_of(&As[0][crow * 32 + ccol]);
  const unsigned lB = lds_addr_of(&Bs[0][crow * 32 + ccol]);

  async_ld_b128(lA,      gA);
  async_ld_b128(lA + 16, gA + 8);
  async_ld_b128(lB,      gB);
  async_ld_b128(lB + 16, gB + 8);

  v8f acc[2][4];
#pragma unroll
  for (int i = 0; i < 2; ++i)
#pragma unroll
    for (int j = 0; j < 4; ++j) acc[i][j] = zero8();

  for (int k = 0; k < C_IN / 32; ++k) {
    const int p = k & 1;
    if (k + 1 < C_IN / 32) {
      const unsigned boff = (unsigned)((k + 1) & 1) * (128 * 32 * 2);
      const bf16_t* nA = gA + (k + 1) * 32;
      const bf16_t* nB = gB + (k + 1) * 32;
      async_ld_b128(lA + boff,      nA);
      async_ld_b128(lA + boff + 16, nA + 8);
      async_ld_b128(lB + boff,      nB);
      async_ld_b128(lB + boff + 16, nB + 8);
      WAIT_ASYNC(4);
    } else {
      WAIT_ASYNC(0);
    }
    __syncthreads();
#pragma unroll
    for (int mf = 0; mf < 2; ++mf) {
      v16bf a = load_a_frag(&As[p][(wm * 32 + mf * 16) * 32], 32, lane);
#pragma unroll
      for (int nf = 0; nf < 4; ++nf) {
        v16bf b = load_b_frag(&Bs[p][(wn * 64 + nf * 16) * 32], 32, lane);
        acc[mf][nf] = __builtin_amdgcn_wmma_f32_16x16x32_bf16(false, a, false, b,
                                                              (short)0, acc[mf][nf], false, false);
      }
    }
    __syncthreads();
  }

  // epilogue: out[(b*512 + col)*16384 + s], tokens contiguous in s per lane -> float4 stores
  const int hi = lane >> 4, cl = lane & 15;
#pragma unroll
  for (int mf = 0; mf < 2; ++mf)
#pragma unroll
    for (int nf = 0; nf < 4; ++nf) {
      const int col  = colBase + wn * 64 + nf * 16 + cl;
      const float bv = bias[col];
      const int row0 = rowBase + wm * 32 + mf * 16 + hi * 8;   // multiple of 8
      const int bidx = row0 >> 14;
      const int s0   = row0 & (S_TOK - 1);
      float* dst = out + ((size_t)bidx * C_IN + col) * S_TOK + s0;
      float4 lo = make_float4(acc[mf][nf][0] + bv, acc[mf][nf][1] + bv,
                              acc[mf][nf][2] + bv, acc[mf][nf][3] + bv);
      float4 hi4 = make_float4(acc[mf][nf][4] + bv, acc[mf][nf][5] + bv,
                               acc[mf][nf][6] + bv, acc[mf][nf][7] + bv);
      *(float4*)(dst)     = lo;
      *(float4*)(dst + 4) = hi4;
    }
}

extern "C" void kernel_launch(void* const* d_in, const int* in_sizes, int n_in,
                              void* d_out, int out_size, void* d_ws, size_t ws_size,
                              hipStream_t stream) {
  (void)in_sizes; (void)n_in; (void)out_size; (void)ws_size;
  const float* x      = (const float*)d_in[0];
  const float* qkv_w  = (const float*)d_in[1];
  const float* qkv_b  = (const float*)d_in[2];
  const float* proj_w = (const float*)d_in[3];
  const float* proj_b = (const float*)d_in[4];
  float* out = (float*)d_out;

  char* ws = (char*)d_ws;
  bf16_t* xt   = (bf16_t*)(ws);                       // 32768*512*2   = 32 MB
  bf16_t* wq   = (bf16_t*)(ws + (size_t)33554432);    // 1536*512*2    = 1.5 MB
  bf16_t* wp   = (bf16_t*)(ws + (size_t)35127296);    // 512*512*2     = 0.5 MB
  bf16_t* qkvb = (bf16_t*)(ws + (size_t)35651584);    // 32768*1536*2  = 96 MB
  bf16_t* aob  = (bf16_t*)(ws + (size_t)136314880);   // 32768*512*2   = 32 MB

  xpose_kernel<<<dim3(S_TOK / 32, C_IN / 32, 2), dim3(32, 8, 1), 0, stream>>>(x, xt);
  cvt_kernel<<<dim3((C_QKV * C_IN + 255) / 256), 256, 0, stream>>>(qkv_w, wq, C_QKV * C_IN);
  cvt_kernel<<<dim3((C_IN * C_IN + 255) / 256), 256, 0, stream>>>(proj_w, wp, C_IN * C_IN);
  qkv_gemm_kernel<<<dim3(C_QKV / 128, N_TOK / 128), 256, 0, stream>>>(xt, wq, qkv_b, qkvb);
  attn_kernel<<<dim3(N_TOK / 2), 64, 0, stream>>>(qkvb, aob);
  proj_gemm_kernel<<<dim3(C_IN / 128, N_TOK / 128), 256, 0, stream>>>(aob, wp, proj_b, out);
}
